// UniLstm_22960895164609
// MI455X (gfx1250) — matrix operand order repaired and checked by reading, not compiled, not run on hardware
//
#include <hip/hip_runtime.h>
#include <hip/hip_bf16.h>

// ---------------------------------------------------------------------------
// UniLstm on MI455X (gfx1250, wave32, WMMA, async LDS staging).
//   Phase 0: convert x -> f16; swizzle Wi/Wh into WMMA-fragment-major f16
//            layout [kblock][ntile][lane][16]; seed h double-buffer; bar=0.
//   Phase 1: ZI[B*T, 4H] = Xf16 @ Wif16 (WMMA GEMM, tiles staged to LDS via
//            global_load_async_to_lds_b128).
//   Phase 2: persistent 32-WG scan; Wh slice LDS-resident; h staged async
//            each step; 16 v_wmma per wave per step; global atomic barrier.
// ---------------------------------------------------------------------------

typedef __attribute__((ext_vector_type(16))) _Float16 v16h;
typedef __attribute__((ext_vector_type(8)))  float    v8f;

#define B_  32
#define T_  2048
#define F_  512
#define H_  512
#define G_  2048          // 4*H
#define NWG_SCAN 32       // scan workgroups; each owns 16 h-cols (64 gate cols)

// ---- CDNA5 async memory->LDS copy (16B per lane, ASYNCcnt-tracked) --------
// Generic LDS pointer: low 32 bits are the LDS byte address (ISA 10.2).
__device__ __forceinline__ void async_copy16(const void* g, void* l) {
  unsigned laddr = (unsigned)(unsigned long long)l;
  unsigned long long ga = (unsigned long long)g;
  asm volatile("global_load_async_to_lds_b128 %0, %1, off"
               :: "v"(laddr), "v"(ga) : "memory");
}
__device__ __forceinline__ void async_wait0() {
  asm volatile("s_wait_asynccnt 0x0" ::: "memory");
}

// ---- WMMA fragment helpers (16-bit layouts per CDNA5 ISA 7.12.2) ----------

// 8 contiguous f16 (16B) + 8 contiguous f16 at +16 elements -> v16h
__device__ __forceinline__ v16h load_row_frag(const _Float16* p) {
  union { v16h v; float4 q[2]; } u;
  u.q[0] = *(const float4*)(p);
  u.q[1] = *(const float4*)(p + 16);
  return u.v;
}

// A matrix 16x32 f16 from row-major LDS (ld = leading dim).
// lane<16: M=lane, K=k0+{0..7,16..23}; lane>=16: M=lane-16, K=k0+{8..15,24..31}
__device__ __forceinline__ v16h load_a_frag(const _Float16* s, int ld,
                                            int row0, int k0, int lane) {
  int m  = row0 + (lane & 15);
  int kb = k0 + ((lane >> 4) << 3);
  return load_row_frag(s + (size_t)m * ld + kb);
}

// B fragments come from the pre-swizzled fragment-major layout:
//   frag(block) = 32 lanes x 16 contiguous f16  (512 elements per block)
__device__ __forceinline__ v16h load_b_swz(const _Float16* blockBase, int lane) {
  union { v16h v; float4 q[2]; } u;
  u.q[0] = *(const float4*)(blockBase + lane * 16);
  u.q[1] = *(const float4*)(blockBase + lane * 16 + 8);
  return u.v;
}

__device__ __forceinline__ v8f wmma_f16(v16h a, v16h b, v8f c) {
  return __builtin_amdgcn_wmma_f32_16x16x32_f16(
      /*neg_a=*/false, a, /*neg_b=*/false, b,
      /*c_mod=*/(short)0, c, /*reuse_a=*/false, /*reuse_b=*/false);
}

// D 16x16 f32: elem r -> row = row0 + r + 8*(lane>=16), col = col0 + (lane&15)
__device__ __forceinline__ void store_d_global(float* Z, size_t ld,
                                               int row0, int col0, int lane, v8f d) {
  int c  = col0 + (lane & 15);
  int r0 = row0 + ((lane & 16) ? 8 : 0);
#pragma unroll
  for (int r = 0; r < 8; ++r) Z[(size_t)(r0 + r) * ld + c] = d[r];
}
__device__ __forceinline__ void store_d_lds(float* Z, int ld,
                                            int row0, int col0, int lane, v8f d) {
  int c  = col0 + (lane & 15);
  int r0 = row0 + ((lane & 16) ? 8 : 0);
#pragma unroll
  for (int r = 0; r < 8; ++r) Z[(r0 + r) * ld + c] = d[r];
}

__device__ __forceinline__ float sigmoidf_(float x) {
  return 1.0f / (1.0f + __expf(-x));
}

// ---------------------------------------------------------------------------
// Phase 0: x convert; Wi/Wh swizzle to fragment-major; hbuf seed; bar init.
// Swizzled layout, elem l: kblock=l>>16; ntile=(l>>9)&127; lane=(l>>4)&31;
// e=l&15;  n = ntile*16 + (lane&15);
// k = kblock*32 + ((lane&16)?8:0) + (e&7) + ((e>=8)?16:0);  src = W[k*G + n]
// ---------------------------------------------------------------------------
__global__ void cvt_kernel(const float* __restrict__ x,
                           const float* __restrict__ Wi,
                           const float* __restrict__ Wh,
                           const float* __restrict__ h0,
                           _Float16* __restrict__ xf,
                           _Float16* __restrict__ Wis,
                           _Float16* __restrict__ Whs_g,
                           _Float16* __restrict__ hbuf,
                           unsigned* __restrict__ bar) {
  size_t tid    = (size_t)blockIdx.x * blockDim.x + threadIdx.x;
  size_t stride = (size_t)gridDim.x * blockDim.x;
  const size_t NX = (size_t)B_ * T_ * F_;        // 33554432
  const size_t NW = (size_t)F_ * G_;             // 1048576
  for (size_t i = tid; i < NX; i += stride) xf[i] = (_Float16)x[i];
  for (size_t l = tid; l < NW; l += stride) {
    int kblock = (int)(l >> 16);
    int ntile  = (int)(l >> 9) & 127;
    int lane   = (int)(l >> 4) & 31;
    int e      = (int)l & 15;
    int n = ntile * 16 + (lane & 15);
    int k = kblock * 32 + ((lane & 16) ? 8 : 0) + (e & 7) + ((e >= 8) ? 16 : 0);
    size_t src = (size_t)k * G_ + n;
    Wis[l]   = (_Float16)Wi[src];
    Whs_g[l] = (_Float16)Wh[src];
  }
  for (size_t i = tid; i < (size_t)B_ * H_; i += stride)
    hbuf[i] = (_Float16)h0[i];                   // seed buffer 0 with h0
  if (tid == 0) *bar = 0u;
}

// ---------------------------------------------------------------------------
// Phase 1: ZI = Xf16 @ Wif16   (M=65536, K=512, N=2048), f32 out.
// 256 thr = 8 waves; WG tile 128x64; wave tile 32x32 (2x2 WMMA frags).
// A tile + 4 B-fragment blocks staged with global_load_async_to_lds_b128.
// ---------------------------------------------------------------------------
__global__ __launch_bounds__(256) void gemm_zi(const _Float16* __restrict__ X,
                                               const _Float16* __restrict__ Wis,
                                               float* __restrict__ Z) {
  __shared__ __align__(16) _Float16 As[128][32];   // 8 KB row-major A tile
  __shared__ __align__(16) _Float16 Bs[4 * 512];   // 4 KB: 4 frag blocks
  const int NT = G_ / 64;                          // 32 N-tiles of 64
  int mt = blockIdx.x / NT, nt = blockIdx.x % NT;
  int m0 = mt * 128, n0 = nt * 64;
  int tid = threadIdx.x, lane = tid & 31, w = tid >> 5;
  int ms = (w >> 1) * 32, ns = (w & 1) * 32;

  v8f a00 = {}, a01 = {}, a10 = {}, a11 = {};

  for (int k0 = 0; k0 < F_; k0 += 32) {
    // A tile: 128 rows x 32 f16 = 512 x 16B chunks (4 chunks per row)
#pragma unroll
    for (int i = 0; i < 2; ++i) {
      int ci = tid + i * 256;
      int r = ci >> 2, c4 = ci & 3;
      async_copy16(X + (size_t)(m0 + r) * F_ + k0 + c4 * 8,
                   (char*)As + (size_t)ci * 16);
    }
    // B: 4 consecutive pre-swizzled frag blocks (2048 f16 = 256 x 16B), flat
    {
      const _Float16* gb = Wis + ((size_t)(k0 >> 5) * 128 + (n0 >> 4)) * 512;
      async_copy16(gb + (size_t)tid * 8, (char*)Bs + (size_t)tid * 16);
    }
    async_wait0();
    __syncthreads();

    v16h fa0 = load_a_frag(&As[0][0], 32, ms, 0, lane);
    v16h fa1 = load_a_frag(&As[0][0], 32, ms + 16, 0, lane);
    v16h fb0 = load_b_swz(Bs + ((ns >> 4) + 0) * 512, lane);
    v16h fb1 = load_b_swz(Bs + ((ns >> 4) + 1) * 512, lane);
    a00 = wmma_f16(fa0, fb0, a00);
    a01 = wmma_f16(fa0, fb1, a01);
    a10 = wmma_f16(fa1, fb0, a10);
    a11 = wmma_f16(fa1, fb1, a11);
    __syncthreads();
  }

  store_d_global(Z, G_, m0 + ms,      n0 + ns,      lane, a00);
  store_d_global(Z, G_, m0 + ms,      n0 + ns + 16, lane, a01);
  store_d_global(Z, G_, m0 + ms + 16, n0 + ns,      lane, a10);
  store_d_global(Z, G_, m0 + ms + 16, n0 + ns + 16, lane, a11);
}

// ---------------------------------------------------------------------------
// Phase 2: persistent scan. 32 WGs x 256 thr. WG wg owns h-cols
// [16*wg, 16*wg+16) -> 64 gate cols. Wh fragment blocks LDS-resident:
// blk = kblock*4 + gate  (kblock 0..15), 512 f16 each.
// ---------------------------------------------------------------------------
__global__ __launch_bounds__(256) void lstm_scan(const float* __restrict__ c0,
                                                 const float* __restrict__ bias,
                                                 const float* __restrict__ zi,
                                                 const _Float16* __restrict__ Whs_g,
                                                 _Float16* __restrict__ hbuf,
                                                 unsigned* __restrict__ bar,
                                                 float* __restrict__ out) {
  __shared__ __align__(16) _Float16 WhsL[64 * 512];  // 64 KB, resident frags
  __shared__ __align__(16) _Float16 Hs[B_ * H_];     // 32 KB, restaged each t
  __shared__ float Outs[B_][64];                     // 8 KB gate pre-acts

  const int tid  = threadIdx.x;
  const int lane = tid & 31;
  const int w    = tid >> 5;                         // 8 waves
  const int wg   = blockIdx.x;                       // 0..31
  const int ncol0 = wg * 16;                         // h-col slice base
  const int m_t = (w >> 2) * 16;                     // batch-row tile {0,16}
  const int n_t = (w & 3) * 16;                      // gate-col tile

  const size_t CF_OFF = 0;
  const size_t HF_OFF = (size_t)B_ * H_;             // 16384
  const size_t YS_OFF = (size_t)2 * B_ * H_;         // 32768

  // Resident Wh fragment blocks, async flat copies of 16B chunks.
  // chunk ci in [0,4096): lds = ci*16B; kblock=ci>>8; gate=(ci>>6)&3;
  // global elem = ((kblock*128) + (gate*32 + wg))*512 + (ci&63)*8
#pragma unroll
  for (int i = 0; i < 16; ++i) {
    int ci = tid + i * 256;
    size_t gidx = ((size_t)(ci >> 8) * 128 + (size_t)((ci >> 6) & 3) * 32 + wg) * 512
                + (size_t)(ci & 63) * 8;
    async_copy16(Whs_g + gidx, (char*)WhsL + (size_t)ci * 16);
  }

  // Fixed thread<->element mapping; c and biases live in registers for all T.
  const int e0 = tid,       b0 = e0 >> 4, hc0 = e0 & 15;
  const int e1 = tid + 256, b1 = e1 >> 4, hc1 = e1 & 15;
  float c_0 = c0[(size_t)b0 * H_ + ncol0 + hc0];
  float c_1 = c0[(size_t)b1 * H_ + ncol0 + hc1];
  const float bi0 = bias[       ncol0 + hc0], bi1 = bias[       ncol0 + hc1];
  const float bf0 = bias[H_   + ncol0 + hc0], bf1 = bias[H_   + ncol0 + hc1];
  const float bg0 = bias[2*H_ + ncol0 + hc0], bg1 = bias[2*H_ + ncol0 + hc1];
  const float bo0 = bias[3*H_ + ncol0 + hc0], bo1 = bias[3*H_ + ncol0 + hc1];
  const float* zb0 = zi + (size_t)b0 * T_ * G_ + ncol0 + hc0;
  const float* zb1 = zi + (size_t)b1 * T_ * G_ + ncol0 + hc1;
  const int hidx0 = b0 * H_ + ncol0 + hc0;
  const int hidx1 = b1 * H_ + ncol0 + hc1;

  async_wait0();
  __syncthreads();   // WhsL ready

  for (int t = 0; t < T_; ++t) {
    const _Float16* hprev = hbuf + (size_t)(t & 1)       * (B_ * H_);
    _Float16*       hnext = hbuf + (size_t)((t + 1) & 1) * (B_ * H_);

    // Stage h_prev [32,512] f16 (2048 x 16B) into LDS, async direct-to-LDS.
#pragma unroll
    for (int i = 0; i < 8; ++i) {
      int ci = tid + i * 256;
      async_copy16((const char*)hprev + (size_t)ci * 16,
                   (char*)Hs + (size_t)ci * 16);
    }
    async_wait0();
    __syncthreads();

    // Prefetch next timestep's zi slice (global_prefetch_b8)
    if ((t + 1) < T_ && hc0 == 0) {
      const float* zn = zb0 + (size_t)(t + 1) * G_;
      __builtin_prefetch(zn,          0, 1);
      __builtin_prefetch(zn + H_,     0, 1);
      __builtin_prefetch(zn + 2 * H_, 0, 1);
      __builtin_prefetch(zn + 3 * H_, 0, 1);
    }

    // out[32,64] = h_prev @ WhSlice : per wave one 16x16 tile, 16 K-steps
    {
      v8f acc = {};
#pragma unroll
      for (int k0 = 0; k0 < F_; k0 += 32) {
        v16h fa = load_a_frag(Hs, H_, m_t, k0, lane);
        v16h fb = load_b_swz(WhsL + ((k0 >> 5) * 4 + (n_t >> 4)) * 512, lane);
        acc = wmma_f16(fa, fb, acc);
      }
      store_d_lds(&Outs[0][0], 64, m_t, n_t, lane, acc);
    }
    __syncthreads();

    // Gate math (i,f,g,o), c/h update, writes.
    const float* z0 = zb0 + (size_t)t * G_;
    const float* z1 = zb1 + (size_t)t * G_;

    float vi0 = Outs[b0][hc0]      + z0[0]      + bi0;
    float vf0 = Outs[b0][16 + hc0] + z0[H_]     + bf0;
    float vg0 = Outs[b0][32 + hc0] + z0[2 * H_] + bg0;
    float vo0 = Outs[b0][48 + hc0] + z0[3 * H_] + bo0;
    float vi1 = Outs[b1][hc1]      + z1[0]      + bi1;
    float vf1 = Outs[b1][16 + hc1] + z1[H_]     + bf1;
    float vg1 = Outs[b1][32 + hc1] + z1[2 * H_] + bg1;
    float vo1 = Outs[b1][48 + hc1] + z1[3 * H_] + bo1;

    c_0 = sigmoidf_(vf0) * c_0 + sigmoidf_(vi0) * tanhf(vg0);
    c_1 = sigmoidf_(vf1) * c_1 + sigmoidf_(vi1) * tanhf(vg1);
    float h_0 = sigmoidf_(vo0) * tanhf(c_0);
    float h_1 = sigmoidf_(vo1) * tanhf(c_1);

    hnext[hidx0] = (_Float16)h_0;
    hnext[hidx1] = (_Float16)h_1;
    out[YS_OFF + ((size_t)b0 * T_ + t) * H_ + ncol0 + hc0] = h_0;
    out[YS_OFF + ((size_t)b1 * T_ + t) * H_ + ncol0 + hc1] = h_1;
    if (t == T_ - 1) {
      out[CF_OFF + hidx0] = c_0;
      out[CF_OFF + hidx1] = c_1;
      out[HF_OFF + hidx0] = h_0;
      out[HF_OFF + hidx1] = h_1;
    }

    // Device-wide split barrier: release-add, acquire-spin, s_sleep while idle
    __threadfence();
    __syncthreads();
    if (tid == 0) {
      __hip_atomic_fetch_add(bar, 1u, __ATOMIC_RELEASE, __HIP_MEMORY_SCOPE_AGENT);
      unsigned target = (unsigned)(t + 1) * NWG_SCAN;
      while (__hip_atomic_load(bar, __ATOMIC_ACQUIRE, __HIP_MEMORY_SCOPE_AGENT) < target) {
        __builtin_amdgcn_s_sleep(1);
      }
    }
    __syncthreads();
  }
}

// ---------------------------------------------------------------------------
extern "C" void kernel_launch(void* const* d_in, const int* in_sizes, int n_in,
                              void* d_out, int out_size, void* d_ws, size_t ws_size,
                              hipStream_t stream) {
  (void)in_sizes; (void)n_in; (void)out_size; (void)ws_size;
  const float* x    = (const float*)d_in[0];   // [32, 2048, 512]
  const float* c0   = (const float*)d_in[1];   // [32, 512]
  const float* h0   = (const float*)d_in[2];   // [32, 512]
  const float* Wi   = (const float*)d_in[3];   // [512, 2048]
  const float* Wh   = (const float*)d_in[4];   // [512, 2048]
  const float* bias = (const float*)d_in[5];   // [2048]
  float* out = (float*)d_out;                  // c_fin | h_fin | ys

  char* ws = (char*)d_ws;
  const size_t SZ_XF  = (size_t)B_ * T_ * F_ * 2;   // 64 MB
  const size_t SZ_WF  = (size_t)F_ * G_ * 2;        // 2 MB each (swizzled)
  const size_t SZ_ZI  = (size_t)B_ * T_ * G_ * 4;   // 512 MB
  const size_t SZ_HB  = (size_t)2 * B_ * H_ * 2;    // 64 KB
  size_t off = 0;
  _Float16* xf    = (_Float16*)(ws + off); off += SZ_XF;
  _Float16* Wis   = (_Float16*)(ws + off); off += SZ_WF;
  _Float16* Whs_g = (_Float16*)(ws + off); off += SZ_WF;
  float*    zi    = (float*)   (ws + off); off += SZ_ZI;
  _Float16* hbuf  = (_Float16*)(ws + off); off += SZ_HB;
  unsigned* bar   = (unsigned*)(ws + off); off += 256;

  // Phase 0: convert + swizzle + init
  hipLaunchKernelGGL(cvt_kernel, dim3(2048), dim3(256), 0, stream,
                     x, Wi, Wh, h0, xf, Wis, Whs_g, hbuf, bar);
  // Phase 1: ZI = X @ Wi  (512 M-tiles x 32 N-tiles)
  hipLaunchKernelGGL(gemm_zi, dim3(512 * 32), dim3(256), 0, stream, xf, Wis, zi);
  // Phase 2: persistent recurrent scan
  hipLaunchKernelGGL(lstm_scan, dim3(NWG_SCAN), dim3(256), 0, stream,
                     c0, bias, zi, Whs_g, hbuf, bar, out);
}
